// CustomLSTM_11192684773522
// MI455X (gfx1250) — matrix operand order repaired and checked by reading, not compiled, not run on hardware
//
#include <hip/hip_runtime.h>

typedef unsigned short u16;
typedef __attribute__((ext_vector_type(16))) __bf16       v16bf;
typedef __attribute__((ext_vector_type(8)))  float        v8f;
typedef __attribute__((ext_vector_type(4)))  unsigned int u32x4;

union V16 { v16bf v; u32x4 q[2]; };

#define WMMA_BF16(A, B, C) \
  __builtin_amdgcn_wmma_f32_16x16x32_bf16(false, (A), false, (B), (short)0, (C), false, false)

__device__ __forceinline__ u16 f2bf(float f) {
  unsigned u = __float_as_uint(f);
  u += 0x7FFFu + ((u >> 16) & 1u);   // round-to-nearest-even
  return (u16)(u >> 16);
}
__device__ __forceinline__ float bf2f(u16 h) {
  return __uint_as_float(((unsigned)h) << 16);
}
__device__ __forceinline__ float sigm(float x) { return 1.f / (1.f + __expf(-x)); }

// CDNA5 async global->LDS copy (16 bytes/lane), tracked by ASYNCcnt.
__device__ __forceinline__ void async_copy_b128(const u16* g, unsigned lds_byte_off) {
  asm volatile("global_load_async_to_lds_b128 %0, %1, off"
               :: "v"(lds_byte_off), "v"(g)
               : "memory");
}
__device__ __forceinline__ void wait_async0() {
#if __has_builtin(__builtin_amdgcn_s_wait_asynccnt)
  __builtin_amdgcn_s_wait_asynccnt(0);
#else
  asm volatile("s_wait_asynccnt 0" ::: "memory");
#endif
}

// ---------------- utility kernels ----------------

__global__ void cvt_f32_bf16(const float* __restrict__ in, u16* __restrict__ out, int n) {
  int i = blockIdx.x * blockDim.x + threadIdx.x;
  if (i < n) out[i] = f2bf(in[i]);
}

__global__ void bias_sum(const float* __restrict__ a, const float* __restrict__ b,
                         float* __restrict__ o, int n) {
  int i = blockIdx.x * blockDim.x + threadIdx.x;
  if (i < n) o[i] = a[i] + b[i];
}

__global__ void zero_hc(u16* __restrict__ h, float* __restrict__ c, int n) {
  int i = blockIdx.x * blockDim.x + threadIdx.x;
  if (i < n) { h[i] = 0; c[i] = 0.f; }
}

// copy final h (= y1[:,511,:]) and c into the output tail
__global__ void finalize(const float* __restrict__ y1, const float* __restrict__ c,
                         float* __restrict__ out_hc) {
  int i = blockIdx.x * blockDim.x + threadIdx.x;
  if (i < 32768) {
    int b = i >> 10, q = i & 1023;
    out_hc[i]         = y1[((size_t)b * 512 + 511) * 1024 + q];
    out_hc[32768 + i] = c[i];
  }
}

// ---------------- hoisted input-projection GEMM ----------------
// P[16384, 4096](bf16) = A[16384, K](bf16) @ W[4096, K]^T(bf16)
// 8 waves / 256-thread block; each wave: one 16-row M tile x 64-col N strip.
__global__ void __launch_bounds__(256) gemm_in(const u16* __restrict__ A,
                                               const u16* __restrict__ W,
                                               u16* __restrict__ P, int K) {
  const int lane = threadIdx.x & 31;
  const int wave = threadIdx.x >> 5;
  const int wg   = blockIdx.x * 8 + wave;   // 65536 waves total
  const int mt   = wg >> 6;                 // 0..1023 M tiles
  const int ns   = wg & 63;                 // 0..63  N strips of 64
  const int sel  = lane >> 4;               // half-wave select
  const int l15  = lane & 15;

  const u16* arow = A + (size_t)(mt * 16 + l15) * K;
  const u16* w0   = W + (size_t)(ns * 64 +  0 + l15) * K;
  const u16* w1   = W + (size_t)(ns * 64 + 16 + l15) * K;
  const u16* w2   = W + (size_t)(ns * 64 + 32 + l15) * K;
  const u16* w3   = W + (size_t)(ns * 64 + 48 + l15) * K;

  v8f acc0 = {}, acc1 = {}, acc2 = {}, acc3 = {};
#pragma unroll 2
  for (int kb = 0; kb < K; kb += 32) {
    V16 a;
    a.q[0] = *(const u32x4*)(arow + kb +     sel * 16);   // K {0..7} / {16..23}
    a.q[1] = *(const u32x4*)(arow + kb + 8 + sel * 16);   // K {8..15} / {24..31}
    v16bf b0 = *(const v16bf*)(w0 + kb + sel * 16);       // 16 contiguous K per lane
    v16bf b1 = *(const v16bf*)(w1 + kb + sel * 16);
    v16bf b2 = *(const v16bf*)(w2 + kb + sel * 16);
    v16bf b3 = *(const v16bf*)(w3 + kb + sel * 16);
    acc0 = WMMA_BF16(a.v, b0, acc0);
    acc1 = WMMA_BF16(a.v, b1, acc1);
    acc2 = WMMA_BF16(a.v, b2, acc2);
    acc3 = WMMA_BF16(a.v, b3, acc3);
  }
  const int rbase = mt * 16 + sel * 8;
#pragma unroll
  for (int r = 0; r < 8; ++r) {
    size_t rowoff = (size_t)(rbase + r) * 4096 + (size_t)ns * 64;
    P[rowoff +  0 + l15] = f2bf(acc0[r]);
    P[rowoff + 16 + l15] = f2bf(acc1[r]);
    P[rowoff + 32 + l15] = f2bf(acc2[r]);
    P[rowoff + 48 + l15] = f2bf(acc3[r]);
  }
}

// ---------------- fused recurrent step ----------------
// gates = pre[:,t,:] + bias + h_in @ w_hh^T ; LSTM cell; h_out + y + c.
// 512 threads = 16 waves: wave w -> (gate = w&3, ksplit = w>>2).
// h tile (16 rows x 1024 bf16) staged once per block into LDS via
// global_load_async_to_lds_b128; gate/K partials meet in LDS for the cell.
#define HPAD 1032   // 1024 + 8 bf16 row padding (bank spread, keeps 16B align)

__global__ void __launch_bounds__(512) lstm_step(const u16*  __restrict__ h_in,   // [32,1024] bf16
                                                 u16*        __restrict__ h_out,  // [32,1024] bf16
                                                 float*      __restrict__ c_buf,  // [32,1024]
                                                 const u16*  __restrict__ w_hh,   // [4096,1024] bf16
                                                 const u16*  __restrict__ pre,    // [B*T,4096] bf16
                                                 const float* __restrict__ bias,  // [4096] (b_ih+b_hh)
                                                 float*      __restrict__ y_f32,  // layer1: d_out, else null
                                                 u16*        __restrict__ y_bf16, // layer0: y0, else null
                                                 int t) {
  const int lane = threadIdx.x & 31;
  const int wave = threadIdx.x >> 5;   // 0..15
  const int gate = wave & 3;           // i,f,g,o
  const int ks   = wave >> 2;          // K split 0..3 (256 K each)
  const int mt   = blockIdx.x & 1;     // batch tile (32 rows -> 2 tiles)
  const int nq   = blockIdx.x >> 1;    // 0..63 q tiles
  const int sel  = lane >> 4;
  const int l15  = lane & 15;
  const int nglb = gate * 1024 + nq * 16 + l15;   // global gate column

  __shared__ __align__(16) u16 hs[16 * HPAD];     // staged h tile
  __shared__ float part[4][4][16][16];            // [gate][ksplit][m][n]

  // ---- async stage: wave w copies h row (mt*16 + w), 4 x 16B per lane ----
  {
    const u16* src = h_in + (size_t)(mt * 16 + wave) * 1024;
    const unsigned dstrow = (unsigned)(size_t)&hs[wave * HPAD];
#pragma unroll
    for (int i = 0; i < 4; ++i) {
      const int col = (lane + i * 32) * 8;        // 8 bf16 = 16B chunks
      async_copy_b128(src + col, dstrow + col * 2);
    }
    wait_async0();
  }
  __syncthreads();

  // ---- accumulators: ksplit 0 seeds with pre + bias ----
  v8f acc0 = {}, acc1 = {};
  if (ks == 0) {
    const float bn = bias[nglb];
#pragma unroll
    for (int r = 0; r < 8; ++r) {
      int b_ = mt * 16 + sel * 8 + r;
      acc0[r] = bf2f(pre[((size_t)b_ * 512 + t) * 4096 + nglb]) + bn;
    }
  }

  const u16* hrow = &hs[(size_t)l15 * HPAD];      // LDS reads -> ds_load_b128
  const u16* wrow = w_hh + (size_t)nglb * 1024;
  const int  kbeg = ks * 256;
#pragma unroll
  for (int kk = 0; kk < 256; kk += 64) {          // two independent acc chains
    const int kb = kbeg + kk;
    V16 a0, a1;
    a0.q[0] = *(const u32x4*)(hrow + kb +      sel * 16);
    a0.q[1] = *(const u32x4*)(hrow + kb +  8 + sel * 16);
    a1.q[0] = *(const u32x4*)(hrow + kb + 32 + sel * 16);
    a1.q[1] = *(const u32x4*)(hrow + kb + 40 + sel * 16);
    v16bf b0 = *(const v16bf*)(wrow + kb +      sel * 16);
    v16bf b1 = *(const v16bf*)(wrow + kb + 32 + sel * 16);
    acc0 = WMMA_BF16(a0.v, b0, acc0);
    acc1 = WMMA_BF16(a1.v, b1, acc1);
  }

#pragma unroll
  for (int r = 0; r < 8; ++r)
    part[gate][ks][sel * 8 + r][l15] = acc0[r] + acc1[r];
  __syncthreads();

  // ---- reduce K splits + LSTM cell: threads 0..255 own one (m, n) each ----
  if (threadIdx.x < 256) {
    const int e = threadIdx.x;
    const int m = e >> 4, n = e & 15;
    float gi = 0.f, gf = 0.f, gg = 0.f, go = 0.f;
#pragma unroll
    for (int k = 0; k < 4; ++k) {
      gi += part[0][k][m][n];
      gf += part[1][k][m][n];
      gg += part[2][k][m][n];
      go += part[3][k][m][n];
    }
    const int b_ = mt * 16 + m;
    const int q  = nq * 16 + n;
    size_t ci = (size_t)b_ * 1024 + q;
    float cn = sigm(gf) * c_buf[ci] + sigm(gi) * tanhf(gg);
    float h  = sigm(go) * tanhf(cn);
    c_buf[ci] = cn;
    h_out[ci] = f2bf(h);
    size_t yi = ((size_t)b_ * 512 + t) * 1024 + q;
    if (y_f32) y_f32[yi] = h;        // layer 1: fp32 straight to d_out
    else       y_bf16[yi] = f2bf(h); // layer 0: bf16 feed for layer-1 GEMM
  }
}

// ---------------- host ----------------

extern "C" void kernel_launch(void* const* d_in, const int* in_sizes, int n_in,
                              void* d_out, int out_size, void* d_ws, size_t ws_size,
                              hipStream_t stream) {
  (void)in_sizes; (void)n_in; (void)out_size; (void)ws_size;
  const float* x    = (const float*)d_in[0];
  const float* wih0 = (const float*)d_in[1];
  const float* whh0 = (const float*)d_in[2];
  const float* bih0 = (const float*)d_in[3];
  const float* bhh0 = (const float*)d_in[4];
  const float* wih1 = (const float*)d_in[5];
  const float* whh1 = (const float*)d_in[6];
  const float* bih1 = (const float*)d_in[7];
  const float* bhh1 = (const float*)d_in[8];
  float* out = (float*)d_out;

  char* ws = (char*)d_ws;
  size_t off = 0;
  auto take = [&](size_t bytes) -> char* {
    char* p = ws + off;
    off = (off + bytes + 255) & ~(size_t)255;
    return p;
  };
  u16*   pre   = (u16*)  take((size_t)16384 * 4096 * 2);  // 128 MB, reused per layer
  u16*   xbf   = (u16*)  take((size_t)16384 * 512  * 2);  // 16 MB
  u16*   y0bf  = (u16*)  take((size_t)16384 * 1024 * 2);  // 32 MB
  u16*   wih0b = (u16*)  take((size_t)4096 * 512  * 2);
  u16*   whh0b = (u16*)  take((size_t)4096 * 1024 * 2);
  u16*   wih1b = (u16*)  take((size_t)4096 * 1024 * 2);
  u16*   whh1b = (u16*)  take((size_t)4096 * 1024 * 2);
  float* bias0 = (float*)take(4096 * 4);
  float* bias1 = (float*)take(4096 * 4);
  u16*   hping = (u16*)  take((size_t)32 * 1024 * 2);
  u16*   hpong = (u16*)  take((size_t)32 * 1024 * 2);
  float* cbuf  = (float*)take((size_t)32 * 1024 * 4);

  // bf16 operand prep
  cvt_f32_bf16<<<(8388608 + 255) / 256, 256, 0, stream>>>(x,    xbf,   8388608);
  cvt_f32_bf16<<<(2097152 + 255) / 256, 256, 0, stream>>>(wih0, wih0b, 2097152);
  cvt_f32_bf16<<<(4194304 + 255) / 256, 256, 0, stream>>>(whh0, whh0b, 4194304);
  cvt_f32_bf16<<<(4194304 + 255) / 256, 256, 0, stream>>>(wih1, wih1b, 4194304);
  cvt_f32_bf16<<<(4194304 + 255) / 256, 256, 0, stream>>>(whh1, whh1b, 4194304);
  bias_sum<<<16, 256, 0, stream>>>(bih0, bhh0, bias0, 4096);
  bias_sum<<<16, 256, 0, stream>>>(bih1, bhh1, bias1, 4096);

  // ----- layer 0 -----
  zero_hc<<<128, 256, 0, stream>>>(hping, cbuf, 32768);
  gemm_in<<<8192, 256, 0, stream>>>(xbf, wih0b, pre, 512);
  for (int t = 0; t < 512; ++t) {
    const u16* hin = (t & 1) ? hpong : hping;
    u16*       hou = (t & 1) ? hping : hpong;
    lstm_step<<<128, 512, 0, stream>>>(hin, hou, cbuf, whh0b, pre, bias0,
                                       nullptr, y0bf, t);
  }

  // ----- layer 1 -----
  zero_hc<<<128, 256, 0, stream>>>(hping, cbuf, 32768);
  gemm_in<<<8192, 256, 0, stream>>>(y0bf, wih1b, pre, 1024);
  for (int t = 0; t < 512; ++t) {
    const u16* hin = (t & 1) ? hpong : hping;
    u16*       hou = (t & 1) ? hping : hpong;
    lstm_step<<<128, 512, 0, stream>>>(hin, hou, cbuf, whh1b, pre, bias1,
                                       out, nullptr, t);
  }

  // tail: h = y1[:,511,:], c
  finalize<<<128, 256, 0, stream>>>(out, cbuf, out + (size_t)16777216);
}
